// SelectionRNN_83837761618688
// MI455X (gfx1250) — matrix operand order repaired and checked by reading, not compiled, not run on hardware
//
#include <hip/hip_runtime.h>
#include <hip/hip_bf16.h>

// ---------------------------------------------------------------------------
// SelectionRNN for MI455X (gfx1250): bf16 WMMA GEMMs, fp32 state, fused gates,
// cross-layer software pipelining (layer0@t + layer1@t-1 per launch).
// ---------------------------------------------------------------------------

typedef __bf16 bf16_t;
typedef __attribute__((ext_vector_type(16))) __bf16        v16bf;
typedef __attribute__((ext_vector_type(8)))  float         v8f;
typedef __attribute__((ext_vector_type(4)))  unsigned int  u32x4;

#define LB  64      // batch
#define TT  256     // time steps
#define II  512     // input size
#define HH  1024    // hidden size
#define OO  256     // output size
#define GG  4096    // 4*H (gates)

struct Frag128x2 { u32x4 lo; u32x4 hi; };
union  FragCast  { Frag128x2 p; v16bf v; };

// A-matrix 16x32 bf16 fragment, contiguous k order chosen here (B matches): lanes 0-15 hold row M=lane,
// K [0..15]; lanes 16-31 hold row M=lane-16, K [16..31].
__device__ __forceinline__ v16bf load_a_frag(const bf16_t* rowptr, int k0, int lane) {
  const int kc = k0 + ((lane & 16) ? 16 : 0);
  FragCast f;
  f.p.lo = *(const u32x4*)(rowptr + kc);
  f.p.hi = *(const u32x4*)(rowptr + kc + 8);
  return f.v;
}
// B-matrix 32x16 bf16, same contiguous k order as A: lane n = lane&15 holds column n; lanes 0-15 K=0..15,
// lanes 16-31 K=16..31 (16 contiguous halves from the weight row, since
// B[k][n] = W[n][k] and W is row-major [4H, K]).
__device__ __forceinline__ v16bf load_b_frag(const bf16_t* rowptr, int k0, int lane) {
  const int kc = k0 + ((lane & 16) ? 16 : 0);
  FragCast f;
  f.p.lo = *(const u32x4*)(rowptr + kc);
  f.p.hi = *(const u32x4*)(rowptr + kc + 8);
  return f.v;
}

__device__ __forceinline__ float sigmoidf_(float x) { return 1.0f / (1.0f + __expf(-x)); }
__device__ __forceinline__ float tanhf_(float x) {
  x = fminf(fmaxf(x, -15.0f), 15.0f);
  const float e = __expf(2.0f * x);
  return (e - 1.0f) / (e + 1.0f);
}

#define WMMA_BF16(A, Bm, C) \
  __builtin_amdgcn_wmma_f32_16x16x32_bf16(false, (A), false, (Bm), (short)0, (C), false, false)

// ---------------------------------------------------------------------------
// fp32 -> bf16 conversion (grid-stride)
// ---------------------------------------------------------------------------
__global__ void k_cvt_bf16(const float* __restrict__ s, bf16_t* __restrict__ d, int n) {
  int i = blockIdx.x * blockDim.x + threadIdx.x;
  const int stride = gridDim.x * blockDim.x;
  for (; i < n; i += stride) d[i] = (bf16_t)s[i];
}

// ---------------------------------------------------------------------------
// Initial weighted-hidden blend + state copies. One block per (layer, batch).
// LDS tree reductions for row min/max.
// ---------------------------------------------------------------------------
__global__ __launch_bounds__(256) void k_init_states(
    const float* __restrict__ clstm, const float* __restrict__ hidden,
    const float* __restrict__ cell, const int* __restrict__ lens,
    float* __restrict__ h_state, float* __restrict__ c_state,
    bf16_t* __restrict__ hbf) {
  const int l = blockIdx.x >> 6;
  const int b = blockIdx.x & 63;
  const int tid = threadIdx.x;
  const float* crow = clstm + (size_t)b * HH;
  const float* hrow = hidden + ((size_t)l * LB + b) * HH;

  float cmn = 3.4e38f, cmx = -3.4e38f, hmn = 3.4e38f, hmx = -3.4e38f;
  for (int j = tid; j < HH; j += 256) {
    const float cv = crow[j], hv = hrow[j];
    cmn = fminf(cmn, cv); cmx = fmaxf(cmx, cv);
    hmn = fminf(hmn, hv); hmx = fmaxf(hmx, hv);
  }
  __shared__ float s0[256], s1[256], s2[256], s3[256];
  s0[tid] = cmn; s1[tid] = cmx; s2[tid] = hmn; s3[tid] = hmx;
  __syncthreads();
  for (int off = 128; off > 0; off >>= 1) {
    if (tid < off) {
      s0[tid] = fminf(s0[tid], s0[tid + off]);
      s1[tid] = fmaxf(s1[tid], s1[tid + off]);
      s2[tid] = fminf(s2[tid], s2[tid + off]);
      s3[tid] = fmaxf(s3[tid], s3[tid + off]);
    }
    __syncthreads();
  }
  cmn = s0[0]; cmx = s1[0]; hmn = s2[0]; hmx = s3[0];
  const float crange = cmx - cmn;
  const float hrange = hmx - hmn;
  const bool  zl = (lens[b] == 0);

  for (int j = tid; j < HH; j += 256) {
    const float cv = crow[j], hv = hrow[j];
    const float s   = (crange > 0.0f) ? (cv - cmn) / crange : cv;
    const float s2v = (hrange > 0.0f) ? (hmn + s * hrange) : s;
    const float cc = zl ? cv : s2v;
    const float wh = 0.5f * hv + 0.5f * cc;   // HIDDEN_WEIGHT = 0.5
    const size_t idx = ((size_t)l * LB + b) * HH + j;
    h_state[idx] = wh;
    hbf[idx]     = (bf16_t)wh;                // phase-0 bf16 h
    c_state[idx] = cell[idx];
  }
}

// ---------------------------------------------------------------------------
// Per-launch pipelined LSTM step: blocks 0..63 run layer0 at time t,
// blocks 64..127 run layer1 at time t-1 (independent within one launch).
// Each workgroup owns one 16-col hidden tile; 4 waves = 4 batch M-tiles;
// each wave keeps i/f/g/o accumulators so the cell update stays in registers
// on the WMMA C layout.
// ---------------------------------------------------------------------------
struct LayerArgs {
  const bf16_t* xbase;   int xstride;   // per-step input rows (bf16)
  const bf16_t* wih;     int Kx;        // input weights [4H, Kx]
  const bf16_t* hbf_in;                 // recurrent input h (bf16, read phase)
  const bf16_t* whh;                    // recurrent weights [4H, H]
  const float*  b_ih;    const float* b_hh;
  float*        h_state; float* c_state; // fp32 persistent state [B,H]
  bf16_t*       hbf_out;                 // bf16 h (write phase)
  bf16_t*       obase;   int ostride;    // masked output rows (bf16)
  int           t;       int active;
};

__global__ __launch_bounds__(128) void k_lstm_pipe(LayerArgs a0, LayerArgs a1,
                                                   const int* __restrict__ lens) {
  const LayerArgs A = (blockIdx.x < 64) ? a0 : a1;
  if (!A.active) return;

  const int lane = threadIdx.x & 31;
  const int wave = threadIdx.x >> 5;
  const int n0   = (blockIdx.x & 63) << 4;
  const int m0   = wave << 4;
  const int nlan = lane & 15;

  v8f acc0 = {0,0,0,0,0,0,0,0};
  v8f acc1 = {0,0,0,0,0,0,0,0};
  v8f acc2 = {0,0,0,0,0,0,0,0};
  v8f acc3 = {0,0,0,0,0,0,0,0};

  // --- input-projection part: K = Kx over wih ---
  {
    const bf16_t* aprow = A.xbase + (size_t)(m0 + nlan) * A.xstride;
    const bf16_t* w0r = A.wih + (size_t)(0 * HH + n0 + nlan) * A.Kx;
    const bf16_t* w1r = A.wih + (size_t)(1 * HH + n0 + nlan) * A.Kx;
    const bf16_t* w2r = A.wih + (size_t)(2 * HH + n0 + nlan) * A.Kx;
    const bf16_t* w3r = A.wih + (size_t)(3 * HH + n0 + nlan) * A.Kx;
    for (int k0 = 0; k0 < A.Kx; k0 += 32) {
      const v16bf a = load_a_frag(aprow, k0, lane);
      acc0 = WMMA_BF16(a, load_b_frag(w0r, k0, lane), acc0);
      acc1 = WMMA_BF16(a, load_b_frag(w1r, k0, lane), acc1);
      acc2 = WMMA_BF16(a, load_b_frag(w2r, k0, lane), acc2);
      acc3 = WMMA_BF16(a, load_b_frag(w3r, k0, lane), acc3);
    }
  }
  // --- recurrent part: K = H over whh ---
  {
    const bf16_t* aprow = A.hbf_in + (size_t)(m0 + nlan) * HH;
    const bf16_t* u0r = A.whh + (size_t)(0 * HH + n0 + nlan) * HH;
    const bf16_t* u1r = A.whh + (size_t)(1 * HH + n0 + nlan) * HH;
    const bf16_t* u2r = A.whh + (size_t)(2 * HH + n0 + nlan) * HH;
    const bf16_t* u3r = A.whh + (size_t)(3 * HH + n0 + nlan) * HH;
    for (int k0 = 0; k0 < HH; k0 += 32) {
      const v16bf a = load_a_frag(aprow, k0, lane);
      acc0 = WMMA_BF16(a, load_b_frag(u0r, k0, lane), acc0);
      acc1 = WMMA_BF16(a, load_b_frag(u1r, k0, lane), acc1);
      acc2 = WMMA_BF16(a, load_b_frag(u2r, k0, lane), acc2);
      acc3 = WMMA_BF16(a, load_b_frag(u3r, k0, lane), acc3);
    }
  }

  // --- cell update on WMMA C layout: lane owns column col, rows M=r(+8) ---
  const int col = n0 + nlan;
  const float bi  = A.b_ih[0 * HH + col] + A.b_hh[0 * HH + col];
  const float bff = A.b_ih[1 * HH + col] + A.b_hh[1 * HH + col];
  const float bg  = A.b_ih[2 * HH + col] + A.b_hh[2 * HH + col];
  const float bo  = A.b_ih[3 * HH + col] + A.b_hh[3 * HH + col];
  const int moff = (lane & 16) ? 8 : 0;
#pragma unroll
  for (int r = 0; r < 8; ++r) {
    const int bgl = m0 + moff + r;
    int lb = lens[bgl]; lb = (lb < 1) ? 1 : lb;        // torch.clamp(min=1)
    const bool valid = (A.t < lb);
    const size_t idx = (size_t)bgl * HH + col;
    const float iv = sigmoidf_(acc0[r] + bi);
    const float fv = sigmoidf_(acc1[r] + bff);
    const float gv = tanhf_  (acc2[r] + bg);
    const float ov = sigmoidf_(acc3[r] + bo);
    const float cold = A.c_state[idx];
    const float cnew = fv * cold + iv * gv;
    const float hnew = ov * tanhf_(cnew);
    const float hold = A.h_state[idx];
    const float hs = valid ? hnew : hold;
    const float cs = valid ? cnew : cold;
    A.h_state[idx] = hs;
    A.c_state[idx] = cs;
    A.hbf_out[idx] = (bf16_t)hs;                        // next-step recurrent input
    A.obase[(size_t)bgl * A.ostride + col] = (bf16_t)(valid ? hnew : 0.0f);
  }
}

// ---------------------------------------------------------------------------
// Head GEMM 1: Y = relu(A @ W.T + bias), bf16 out. Grid (M/64, N/16), 4 waves.
// ---------------------------------------------------------------------------
__global__ __launch_bounds__(128) void k_gemm_relu_bf16(
    const bf16_t* __restrict__ A, const bf16_t* __restrict__ W,
    const float* __restrict__ bias, bf16_t* __restrict__ Y, int K, int N) {
  const int lane = threadIdx.x & 31;
  const int wave = threadIdx.x >> 5;
  const int m0 = (blockIdx.x * 4 + wave) << 4;
  const int n0 = blockIdx.y << 4;
  const int nlan = lane & 15;

  v8f acc = {0,0,0,0,0,0,0,0};
  const bf16_t* aprow = A + (size_t)(m0 + nlan) * K;
  const bf16_t* wrow  = W + (size_t)(n0 + nlan) * K;
  for (int k0 = 0; k0 < K; k0 += 32) {
    acc = WMMA_BF16(load_a_frag(aprow, k0, lane), load_b_frag(wrow, k0, lane), acc);
  }
  const int col = n0 + nlan;
  const float bv = bias[col];
  const int moff = (lane & 16) ? 8 : 0;
#pragma unroll
  for (int r = 0; r < 8; ++r) {
    const size_t row = (size_t)(m0 + moff + r);
    Y[row * N + col] = (bf16_t)fmaxf(acc[r] + bv, 0.0f);
  }
}

// ---------------------------------------------------------------------------
// Head GEMM 2: logits = relu(A @ W.T + bias) * (len!=0), fp32 out.
// ---------------------------------------------------------------------------
__global__ __launch_bounds__(128) void k_gemm_relu_f32_mask(
    const bf16_t* __restrict__ A, const bf16_t* __restrict__ W,
    const float* __restrict__ bias, float* __restrict__ Out,
    int K, int N, const int* __restrict__ lens) {
  const int lane = threadIdx.x & 31;
  const int wave = threadIdx.x >> 5;
  const int m0 = (blockIdx.x * 4 + wave) << 4;
  const int n0 = blockIdx.y << 4;
  const int nlan = lane & 15;

  v8f acc = {0,0,0,0,0,0,0,0};
  const bf16_t* aprow = A + (size_t)(m0 + nlan) * K;
  const bf16_t* wrow  = W + (size_t)(n0 + nlan) * K;
  for (int k0 = 0; k0 < K; k0 += 32) {
    acc = WMMA_BF16(load_a_frag(aprow, k0, lane), load_b_frag(wrow, k0, lane), acc);
  }
  const int col = n0 + nlan;
  const float bv = bias[col];
  const int moff = (lane & 16) ? 8 : 0;
#pragma unroll
  for (int r = 0; r < 8; ++r) {
    const int row = m0 + moff + r;
    const int b   = row >> 8;                 // row = b*T + t, T = 256
    const float m = (lens[b] == 0) ? 0.0f : 1.0f;
    Out[(size_t)row * N + col] = fmaxf(acc[r] + bv, 0.0f) * m;
  }
}

// ---------------------------------------------------------------------------
// Final hn / cn with len==0 masking.
// ---------------------------------------------------------------------------
__global__ __launch_bounds__(256) void k_final_states(
    const float* __restrict__ hs, const float* __restrict__ cs,
    const int* __restrict__ lens, float* __restrict__ on, float* __restrict__ oc) {
  const int i = blockIdx.x * 256 + threadIdx.x;
  if (i >= 2 * LB * HH) return;
  const int b = (i >> 10) & 63;
  const float m = (lens[b] == 0) ? 0.0f : 1.0f;
  on[i] = hs[i] * m;
  oc[i] = cs[i] * m;
}

// ---------------------------------------------------------------------------
extern "C" void kernel_launch(void* const* d_in, const int* in_sizes, int n_in,
                              void* d_out, int out_size, void* d_ws, size_t ws_size,
                              hipStream_t stream) {
  const float* x     = (const float*)d_in[0];
  const float* clstm = (const float*)d_in[1];
  const float* hid   = (const float*)d_in[2];
  const float* cel   = (const float*)d_in[3];
  const int*   lens  = (const int*)d_in[4];
  const float* w_ih0 = (const float*)d_in[5];
  const float* w_hh0 = (const float*)d_in[6];
  const float* b_ih0 = (const float*)d_in[7];
  const float* b_hh0 = (const float*)d_in[8];
  const float* w_ih1 = (const float*)d_in[9];
  const float* w_hh1 = (const float*)d_in[10];
  const float* b_ih1 = (const float*)d_in[11];
  const float* b_hh1 = (const float*)d_in[12];
  const float* w1    = (const float*)d_in[13];
  const float* b1    = (const float*)d_in[14];
  const float* w2    = (const float*)d_in[15];
  const float* b2    = (const float*)d_in[16];
  (void)in_sizes; (void)n_in; (void)out_size; (void)ws_size;

  // ---- workspace layout (~117 MB) ----
  bf16_t* p = (bf16_t*)d_ws;
  bf16_t* wih0b = p; p += (size_t)GG * II;
  bf16_t* whh0b = p; p += (size_t)GG * HH;
  bf16_t* wih1b = p; p += (size_t)GG * HH;
  bf16_t* whh1b = p; p += (size_t)GG * HH;
  bf16_t* w1b   = p; p += (size_t)HH * HH;
  bf16_t* w2b   = p; p += (size_t)OO * HH;
  bf16_t* xbf   = p; p += (size_t)LB * TT * II;
  bf16_t* lob   = p; p += (size_t)LB * TT * HH;    // masked top-layer outputs
  bf16_t* y1b   = p; p += (size_t)LB * TT * HH;    // head hidden (bf16)
  bf16_t* hbfB  = p; p += (size_t)2 * 2 * LB * HH; // [phase][layer][B][H]
  bf16_t* inpb  = p; p += (size_t)2 * LB * HH;     // layer-1 input, double-buffered
  float*  f = (float*)p;
  float* h_state = f; f += (size_t)2 * LB * HH;    // [L][B][H] fp32
  float* c_state = f; f += (size_t)2 * LB * HH;

  auto cvt = [&](const float* s, bf16_t* d, size_t n) {
    int blocks = (int)((n + 1023) / 1024);
    if (blocks > 2048) blocks = 2048;
    k_cvt_bf16<<<blocks, 256, 0, stream>>>(s, d, (int)n);
  };
  cvt(w_ih0, wih0b, (size_t)GG * II);
  cvt(w_hh0, whh0b, (size_t)GG * HH);
  cvt(w_ih1, wih1b, (size_t)GG * HH);
  cvt(w_hh1, whh1b, (size_t)GG * HH);
  cvt(w1,    w1b,   (size_t)HH * HH);
  cvt(w2,    w2b,   (size_t)OO * HH);
  cvt(x,     xbf,   (size_t)LB * TT * II);

  k_init_states<<<2 * LB, 256, 0, stream>>>(clstm, hid, cel, lens,
                                            h_state, c_state, hbfB);

  // ---- pipelined recurrence: one launch runs layer0@t and layer1@(t-1) ----
  for (int t = 0; t <= TT; ++t) {
    LayerArgs a0 = {}, a1 = {};

    // layer 0 at time t
    a0.active = (t < TT) ? 1 : 0;
    if (a0.active) {
      const int rp = t & 1, wp = rp ^ 1;
      a0.xbase   = xbf + (size_t)t * II;          // row b at +b*(T*I)
      a0.xstride = TT * II;
      a0.wih     = wih0b;  a0.Kx = II;
      a0.hbf_in  = hbfB + (size_t)rp * 2 * LB * HH;          // layer0, read phase
      a0.whh     = whh0b;
      a0.b_ih    = b_ih0;  a0.b_hh = b_hh0;
      a0.h_state = h_state;
      a0.c_state = c_state;
      a0.hbf_out = hbfB + (size_t)wp * 2 * LB * HH;          // layer0, write phase
      a0.obase   = inpb + (size_t)(t & 1) * LB * HH;         // inp slot t&1
      a0.ostride = HH;
      a0.t       = t;
    }

    // layer 1 at time t-1
    const int t1 = t - 1;
    a1.active = (t1 >= 0) ? 1 : 0;
    if (a1.active) {
      const int rp = t1 & 1, wp = rp ^ 1;
      a1.xbase   = inpb + (size_t)(t1 & 1) * LB * HH;        // inp slot (t-1)&1
      a1.xstride = HH;
      a1.wih     = wih1b;  a1.Kx = HH;
      a1.hbf_in  = hbfB + (size_t)rp * 2 * LB * HH + (size_t)LB * HH; // layer1
      a1.whh     = whh1b;
      a1.b_ih    = b_ih1;  a1.b_hh = b_hh1;
      a1.h_state = h_state + (size_t)LB * HH;
      a1.c_state = c_state + (size_t)LB * HH;
      a1.hbf_out = hbfB + (size_t)wp * 2 * LB * HH + (size_t)LB * HH;
      a1.obase   = lob + (size_t)t1 * HH;                    // lstm_out row t-1
      a1.ostride = TT * HH;
      a1.t       = t1;
    }

    k_lstm_pipe<<<128, 128, 0, stream>>>(a0, a1, lens);
  }

  // ---- head ----
  k_gemm_relu_bf16<<<dim3(LB * TT / 64, HH / 16), 128, 0, stream>>>(
      lob, w1b, b1, y1b, HH, HH);
  float* out = (float*)d_out;
  k_gemm_relu_f32_mask<<<dim3(LB * TT / 64, OO / 16), 128, 0, stream>>>(
      y1b, w2b, b2, out, HH, OO, lens);

  // ---- hn / cn ----
  k_final_states<<<(2 * LB * HH) / 256, 256, 0, stream>>>(
      h_state, c_state, lens,
      out + (size_t)LB * TT * OO,
      out + (size_t)LB * TT * OO + 2 * LB * HH);
}
